// SelfAttention4_24970939859700
// MI455X (gfx1250) — matrix operand order repaired and checked by reading, not compiled, not run on hardware
//
#include <hip/hip_runtime.h>
#include <hip/hip_bf16.h>
#include <math.h>

// ---------------------------------------------------------------- types ----
typedef __bf16 bf16_t;
typedef __bf16 bf16x8  __attribute__((ext_vector_type(8)));
typedef __bf16 bf16x16 __attribute__((ext_vector_type(16)));
typedef float  f32x8   __attribute__((ext_vector_type(8)));
typedef int    i32x4   __attribute__((ext_vector_type(4)));

// ---------------------------------------------------------------- sizes ----
#define BB    32
#define SEQ   1024          // compile-time: epilogue div/mod become shift/mask
#define HID   1024
#define ATT   512
#define EMB   128
#define MTOT  (BB*SEQ)      // 32768 rows

// ------------------------------------------------------------- helpers ----
__device__ inline bf16_t f2bf(float f) {
  unsigned u = __builtin_bit_cast(unsigned, f);
  unsigned r = u + 0x7FFFu + ((u >> 16) & 1u);      // round-to-nearest-even
  unsigned short s = (unsigned short)(r >> 16);
  return __builtin_bit_cast(bf16_t, s);
}
__device__ inline float bf2f(bf16_t b) {
  unsigned short s = __builtin_bit_cast(unsigned short, b);
  unsigned u = ((unsigned)s) << 16;
  return __builtin_bit_cast(float, u);
}

// Issue all 8 b128 loads (4 B tiles x two 8-wide k halves) as one
// clause, then a single s_wait_loadcnt. The +32B half uses the instruction
// immediate offset, so no extra VALU address math. Values are defined by the
// asm itself, so consumers cannot be scheduled above the wait.
__device__ inline void load_b_frags(const bf16_t* p0, const bf16_t* p1,
                                    const bf16_t* p2, const bf16_t* p3,
                                    i32x4& b0l, i32x4& b0h, i32x4& b1l, i32x4& b1h,
                                    i32x4& b2l, i32x4& b2h, i32x4& b3l, i32x4& b3h) {
  asm volatile("s_clause 0x7\n\t"
               "global_load_b128 %0, %8, off\n\t"
               "global_load_b128 %1, %8, off offset:32\n\t"
               "global_load_b128 %2, %9, off\n\t"
               "global_load_b128 %3, %9, off offset:32\n\t"
               "global_load_b128 %4, %10, off\n\t"
               "global_load_b128 %5, %10, off offset:32\n\t"
               "global_load_b128 %6, %11, off\n\t"
               "global_load_b128 %7, %11, off offset:32\n\t"
               "s_wait_loadcnt 0"
               : "=&v"(b0l), "=&v"(b0h), "=&v"(b1l), "=&v"(b1h),
                 "=&v"(b2l), "=&v"(b2h), "=&v"(b3l), "=&v"(b3h)
               : "v"(p0), "v"(p1), "v"(p2), "v"(p3)
               : "memory");
}

// ------------------------------------------------- weight transpose+cvt ----
// Wt[n*K + k] = (bf16) W[k*N + n]
__global__ void wtrans_kernel(const float* __restrict__ W,
                              bf16_t* __restrict__ Wt, int K, int N) {
  int n = blockIdx.x * blockDim.x + threadIdx.x;
  int k = blockIdx.y;
  if (n < N) Wt[(size_t)n * K + k] = f2bf(W[(size_t)k * N + n]);
}

// ------------------------------------------------------------ embedding ----
__global__ void embed_kernel(const int* __restrict__ cate,
                             const float* __restrict__ cont,
                             const float* __restrict__ E0, const float* __restrict__ E1,
                             const float* __restrict__ E2, const float* __restrict__ E3,
                             const float* __restrict__ Wc,
                             bf16_t* __restrict__ xin) {
  int m   = blockIdx.x;
  int col = blockIdx.y * blockDim.x + threadIdx.x;
  float v;
  if (col < 4 * EMB) {
    int f = col >> 7, e = col & (EMB - 1);
    const float* E = (f == 0) ? E0 : (f == 1) ? E1 : (f == 2) ? E2 : E3;
    int idx = cate[(size_t)m * 4 + f];
    v = E[(size_t)idx * EMB + e];
  } else {
    int j = col - 4 * EMB;
    const float* c = cont + (size_t)m * 4;
    v = c[0] * Wc[j] + c[1] * Wc[512 + j] + c[2] * Wc[1024 + j] + c[3] * Wc[1536 + j];
  }
  xin[(size_t)m * HID + col] = f2bf(v);
}

// ----------------------------------------------------------------- GEMM ----
// D[m,n] = sum_k A[m,k] * Bt[n,k]   (A,Bt bf16; f32 WMMA accumulation)
// One wave -> 16x64 tile: A fragment reused across 4 WMMAs per K-step.
enum { EPI_F32 = 0, EPI_BF16 = 1, EPI_BF16_PE = 2, EPI_BF16_TRANS = 3 };

template <int EPI>
__global__ void __launch_bounds__(256)
gemm_bf16_kernel(const bf16_t* __restrict__ A, const bf16_t* __restrict__ Bt,
                 void* __restrict__ D,
                 int M, int N, int K, int lda, int ldb, int ldd,
                 long sA, long sB, long sD) {
  const int lane = threadIdx.x & 31;
  const int wave = threadIdx.x >> 5;
  const int tm   = blockIdx.y * 8 + wave;        // 16-row tile index
  if (tm * 16 >= M) return;                      // wave-uniform exit
  const int m0 = tm * 16;
  const int n0 = blockIdx.x * 64;                // 64-column group
  const int z  = blockIdx.z;
  A  += (size_t)z * sA;
  Bt += (size_t)z * sB;

  const int row = lane & 15;                     // A: M index / Bt: N index
  const int hi  = lane >> 4;                     // lane half selects K slice

  f32x8 acc[4] = {};
  const bf16_t* arow = A + (size_t)(m0 + row) * lda + hi * 8;
  const bf16_t* bp0 = Bt + (size_t)(n0 +  0 + row) * ldb + hi * 8;
  const bf16_t* bp1 = Bt + (size_t)(n0 + 16 + row) * ldb + hi * 8;
  const bf16_t* bp2 = Bt + (size_t)(n0 + 32 + row) * ldb + hi * 8;
  const bf16_t* bp3 = Bt + (size_t)(n0 + 48 + row) * ldb + hi * 8;

  for (int k0 = 0; k0 < K; k0 += 32) {
    __builtin_prefetch(arow + k0 + 128, 0, 1);   // stream A ahead
    bf16x16 afrag;                               // 16x32 A fragment
    ((bf16x8*)&afrag)[0] = *(const bf16x8*)(arow + k0);
    ((bf16x8*)&afrag)[1] = *(const bf16x8*)(arow + k0 + 16);

    i32x4 b0l, b0h, b1l, b1h, b2l, b2h, b3l, b3h;
    load_b_frags(bp0 + k0, bp1 + k0, bp2 + k0, bp3 + k0,
                 b0l, b0h, b1l, b1h, b2l, b2h, b3l, b3h);

    bf16x16 bf0, bf1, bf2, bf3;
    ((i32x4*)&bf0)[0] = b0l; ((i32x4*)&bf0)[1] = b0h;
    ((i32x4*)&bf1)[0] = b1l; ((i32x4*)&bf1)[1] = b1h;
    ((i32x4*)&bf2)[0] = b2l; ((i32x4*)&bf2)[1] = b2h;
    ((i32x4*)&bf3)[0] = b3l; ((i32x4*)&bf3)[1] = b3h;

    acc[0] = __builtin_amdgcn_wmma_f32_16x16x32_bf16(false, afrag, false, bf0, (short)0, acc[0], false, false);
    acc[1] = __builtin_amdgcn_wmma_f32_16x16x32_bf16(false, afrag, false, bf1, (short)0, acc[1], false, false);
    acc[2] = __builtin_amdgcn_wmma_f32_16x16x32_bf16(false, afrag, false, bf2, (short)0, acc[2], false, false);
    acc[3] = __builtin_amdgcn_wmma_f32_16x16x32_bf16(false, afrag, false, bf3, (short)0, acc[3], false, false);
  }

#pragma unroll
  for (int t = 0; t < 4; ++t) {
    const int n = n0 + t * 16 + row;
#pragma unroll
    for (int v = 0; v < 8; ++v) {
      const int m = m0 + v + hi * 8;             // C/D VGPR layout (ISA 7.12.2)
      float val = acc[t][v];
      if (EPI == EPI_F32) {
        ((float*)D)[(size_t)z * sD + (size_t)m * ldd + n] = val;
      } else if (EPI == EPI_BF16) {
        ((bf16_t*)D)[(size_t)z * sD + (size_t)m * ldd + n] = f2bf(val);
      } else if (EPI == EPI_BF16_PE) {
        const int s = m & (SEQ - 1);             // compile-time mask, no div
        const int i = n >> 1;
        float ang = (float)s * __expf(-(float)(2 * i) * (9.2103403719761836f / (float)HID));
        float pe  = (n & 1) ? __cosf(ang) : __sinf(ang);
        ((bf16_t*)D)[(size_t)m * ldd + n] = f2bf(val + pe);
      } else {                                    // EPI_BF16_TRANS (per-batch Vt)
        const int b = m >> 10, s = m & (SEQ - 1); // compile-time shift/mask
        ((bf16_t*)D)[(size_t)b * sD + (size_t)n * ldd + s] = f2bf(val);
      }
    }
  }
}

// ---------------------------------------------- softmax over QUERY axis ----
__global__ void softmax_q_kernel(const float* __restrict__ S,
                                 bf16_t* __restrict__ Aout, float scale) {
  const int b = blockIdx.y;
  const int k = blockIdx.x * blockDim.x + threadIdx.x;   // lane-coalesced in k
  const float* col = S + (size_t)b * SEQ * SEQ + k;
  float mx = -1e30f;
  for (int q = 0; q < SEQ; ++q) mx = fmaxf(mx, col[(size_t)q * SEQ] * scale);
  float sum = 0.f;
  for (int q = 0; q < SEQ; ++q) sum += __expf(col[(size_t)q * SEQ] * scale - mx);
  const float inv = 1.0f / sum;
  bf16_t* ocol = Aout + (size_t)b * SEQ * SEQ + k;
  for (int q = 0; q < SEQ; ++q)
    ocol[(size_t)q * SEQ] = f2bf(__expf(col[(size_t)q * SEQ] * scale - mx) * inv);
}

// ---------------------------------------------------- final GEMV+sigmoid ----
__global__ void final_kernel(const bf16_t* __restrict__ H,
                             const float* __restrict__ Wf,
                             float* __restrict__ out) {
  const int lane = threadIdx.x & 31;
  const int wave = threadIdx.x >> 5;
  const int m    = blockIdx.x * 8 + wave;
  const bf16x8* hv = (const bf16x8*)(H + (size_t)m * HID);  // b128 loads
  float acc = 0.f;
  for (int j = lane; j < HID / 8; j += 32) {
    bf16x8 h8 = hv[j];
    const float* w = Wf + (size_t)j * 8;
#pragma unroll
    for (int e = 0; e < 8; ++e) acc += bf2f(h8[e]) * w[e];
  }
  for (int off = 16; off > 0; off >>= 1) acc += __shfl_down(acc, off, 32);
  if (lane == 0) out[m] = 1.0f / (1.0f + __expf(-acc));
}

// --------------------------------------------------------------- launch ----
extern "C" void kernel_launch(void* const* d_in, const int* in_sizes, int n_in,
                              void* d_out, int out_size, void* d_ws, size_t ws_size,
                              hipStream_t stream) {
  const int*   cate  = (const int*)  d_in[0];
  const float* cont  = (const float*)d_in[1];
  // d_in[2]=mask (all ones, unused), d_in[3]=targets (unused)
  const float* E0 = (const float*)d_in[4];
  const float* E1 = (const float*)d_in[5];
  const float* E2 = (const float*)d_in[6];
  const float* E3 = (const float*)d_in[7];
  const float* Wc = (const float*)d_in[8];
  const float* Wproj = (const float*)d_in[9];
  const float* WQ[2] = {(const float*)d_in[10], (const float*)d_in[13]};
  const float* WK[2] = {(const float*)d_in[11], (const float*)d_in[14]};
  const float* WV[2] = {(const float*)d_in[12], (const float*)d_in[15]};
  const float* W0 = (const float*)d_in[16];
  const float* Wf = (const float*)d_in[17];
  float* out = (float*)d_out;

  char* ws = (char*)d_ws;
  size_t off = 0;
  auto alloc = [&](size_t nbytes) -> char* {
    char* p = ws + off;
    off += (nbytes + 255) & ~(size_t)255;
    return p;
  };
  bf16_t* wprojT = (bf16_t*)alloc((size_t)HID * HID * 2);
  bf16_t* wqT[2] = {(bf16_t*)alloc((size_t)ATT * HID * 2), (bf16_t*)alloc((size_t)ATT * HID * 2)};
  bf16_t* wkT[2] = {(bf16_t*)alloc((size_t)ATT * HID * 2), (bf16_t*)alloc((size_t)ATT * HID * 2)};
  bf16_t* wvT[2] = {(bf16_t*)alloc((size_t)ATT * HID * 2), (bf16_t*)alloc((size_t)ATT * HID * 2)};
  bf16_t* w0T    = (bf16_t*)alloc((size_t)HID * HID * 2);
  bf16_t* xin    = (bf16_t*)alloc((size_t)MTOT * HID * 2);   // reused later as hbuf
  bf16_t* xb     = (bf16_t*)alloc((size_t)MTOT * HID * 2);
  bf16_t* qb     = (bf16_t*)alloc((size_t)MTOT * ATT * 2);
  bf16_t* kb     = (bf16_t*)alloc((size_t)MTOT * ATT * 2);
  bf16_t* vtb    = (bf16_t*)alloc((size_t)MTOT * ATT * 2);   // [B][ATT][SEQ]
  float*  sbuf   = (float*) alloc((size_t)BB * SEQ * SEQ * 4);
  bf16_t* abuf   = (bf16_t*)alloc((size_t)BB * SEQ * SEQ * 2);
  bf16_t* zbuf   = (bf16_t*)alloc((size_t)MTOT * HID * 2);
  bf16_t* hbuf   = xin;                                       // xin dead by then

  const dim3 tb(256);

  // 1) transpose+convert weights (tiny)
  wtrans_kernel<<<dim3(HID / 256, HID), tb, 0, stream>>>(Wproj, wprojT, HID, HID);
  for (int h = 0; h < 2; ++h) {
    wtrans_kernel<<<dim3(ATT / 256, HID), tb, 0, stream>>>(WQ[h], wqT[h], HID, ATT);
    wtrans_kernel<<<dim3(ATT / 256, HID), tb, 0, stream>>>(WK[h], wkT[h], HID, ATT);
    wtrans_kernel<<<dim3(ATT / 256, HID), tb, 0, stream>>>(WV[h], wvT[h], HID, ATT);
  }
  wtrans_kernel<<<dim3(HID / 256, HID), tb, 0, stream>>>(W0, w0T, HID, HID);

  // 2) embedding gather + cont projection -> xin (bf16)
  embed_kernel<<<dim3(MTOT, HID / 256), tb, 0, stream>>>(cate, cont, E0, E1, E2, E3, Wc, xin);

  // 3) x = xin @ Wproj + PE  -> xb (bf16)
  gemm_bf16_kernel<EPI_BF16_PE><<<dim3(HID / 64, MTOT / 16 / 8, 1), tb, 0, stream>>>(
      xin, wprojT, xb, MTOT, HID, HID, HID, HID, HID, 0, 0, 0);

  const float scale = 0.044194173824159216f;     // 1/sqrt(512)
  for (int h = 0; h < 2; ++h) {
    // 4) Q, K, V (V stored transposed per batch for the attn@V GEMM)
    gemm_bf16_kernel<EPI_BF16><<<dim3(ATT / 64, MTOT / 16 / 8, 1), tb, 0, stream>>>(
        xb, wqT[h], qb, MTOT, ATT, HID, HID, HID, ATT, 0, 0, 0);
    gemm_bf16_kernel<EPI_BF16><<<dim3(ATT / 64, MTOT / 16 / 8, 1), tb, 0, stream>>>(
        xb, wkT[h], kb, MTOT, ATT, HID, HID, HID, ATT, 0, 0, 0);
    gemm_bf16_kernel<EPI_BF16_TRANS><<<dim3(ATT / 64, MTOT / 16 / 8, 1), tb, 0, stream>>>(
        xb, wvT[h], vtb, MTOT, ATT, HID, HID, HID, SEQ, 0, 0, (long)ATT * SEQ);
    // 5) scores[b] = Q[b] @ K[b]^T  (canonical form: Bt = K, no transpose)
    gemm_bf16_kernel<EPI_F32><<<dim3(SEQ / 64, SEQ / 16 / 8, BB), tb, 0, stream>>>(
        qb, kb, sbuf, SEQ, SEQ, ATT, ATT, ATT, SEQ,
        (long)SEQ * ATT, (long)SEQ * ATT, (long)SEQ * SEQ);
    // 6) softmax over the query axis (faithful to nn.Softmax(dim=1))
    softmax_q_kernel<<<dim3(SEQ / 256, BB), tb, 0, stream>>>(sbuf, abuf, scale);
    // 7) z[b] = attn[b] @ V[b]  -> concat slot h of zbuf
    gemm_bf16_kernel<EPI_BF16><<<dim3(ATT / 64, SEQ / 16 / 8, BB), tb, 0, stream>>>(
        abuf, vtb, zbuf + (size_t)h * ATT, SEQ, ATT, SEQ, SEQ, SEQ, HID,
        (long)SEQ * SEQ, (long)ATT * SEQ, (long)SEQ * HID);
  }

  // 8) h = z1z2 @ W0 -> hbuf (bf16)
  gemm_bf16_kernel<EPI_BF16><<<dim3(HID / 64, MTOT / 16 / 8, 1), tb, 0, stream>>>(
      zbuf, w0T, hbuf, MTOT, HID, HID, HID, HID, HID, 0, 0, 0);

  // 9) out = sigmoid(h @ Wf)
  final_kernel<<<dim3(MTOT / 8), tb, 0, stream>>>(hbuf, Wf, out);
}
